// GPT_58737972740379
// MI455X (gfx1250) — matrix operand order repaired and checked by reading, not compiled, not run on hardware
//
#include <hip/hip_runtime.h>
#include <cmath>

// Problem constants (match reference)
#define B_     2
#define T_     2048
#define C_     1024
#define NH_    16
#define HS_    64
#define BD_    16
#define DELTA_ 32
#define M_     (B_ * T_)          // 4096 rows for all big GEMMs

typedef __attribute__((ext_vector_type(16))) __bf16 v16bf;
typedef __attribute__((ext_vector_type(8)))  float  v8f;
typedef __attribute__((ext_vector_type(4)))  float  v4f;

__device__ __forceinline__ unsigned bf_bits(float f) {
    // round-to-nearest-even f32 -> bf16, result in low 16 bits
    unsigned u = __builtin_bit_cast(unsigned, f);
    return (u + 0x7FFFu + ((u >> 16) & 1u)) >> 16;
}
__device__ __forceinline__ __bf16 f2bf(float f) {
    unsigned short us = (unsigned short)bf_bits(f);
    return __builtin_bit_cast(__bf16, us);
}
__device__ __forceinline__ float gelu_erf(float x) {
    return 0.5f * x * (1.0f + erff(x * 0.70710678118654752440f));
}

// ---------------------------------------------------------------------------
// O[M,N] = X[M,K] @ W[K,N] + bias[N]   (f32 in/out, bf16 WMMA core)
//
// Block = 256 threads = 8 waves as 2(M) x 4(N). Block tile 32M x 256N,
// wave tile 16M x 64N (4 accumulators sharing one A fragment).
// A/B staged in LDS in *fragment order* ([lane][e], k = 16*(lane/16) + e) so fragment
// builds are pure ds_load_b128. Double-buffered: iteration `it` stages tile
// it+1 (global_load_b128 + one-time bf16 conversion) underneath the WMMA
// clause of tile it; a single barrier per K step both publishes the staged
// buffer and protects the buffer being overwritten.
// grid.x = M/32, grid.y = N/256, K multiple of 32.
// ---------------------------------------------------------------------------
__global__ __launch_bounds__(256) void wmma_gemm_bias(
        const float* __restrict__ X,
        const float* __restrict__ W,
        const float* __restrict__ bias,
        float* __restrict__ O,
        int M, int N, int K)
{
    //  Afrag[buf][wm][l][e] : A 16x32 bf16 fragment, row block wm
    //    lane l<16 : row=l,    e 0..15 -> K 0..15
    //    lane l>=16: row=l-16, e 0..15 -> K 16..31
    //  Bfrag[buf][nb][l][e] : B 32x16 bf16 fragment, col block nb (16 cols)
    //    lane l: col n=l&15, e -> K = e + (l>=16 ? 16 : 0)
    __shared__ alignas(32) __bf16 Afrag[2][2][32][16];    //  4 KB
    __shared__ alignas(32) __bf16 Bfrag[2][16][32][16];   // 32 KB

    const int tid  = threadIdx.x;
    const int lane = tid & 31;
    const int wave = tid >> 5;
    const int wm   = wave & 1;          // 0..1 : M sub-block
    const int wn   = wave >> 1;         // 0..3 : N sub-block (64 cols)

    const int row0  = blockIdx.x * 32;
    const int colB0 = blockIdx.y * 256;

    const int  n  = lane & 15;
    const bool hi = lane >= 16;

    v8f acc[4];
    #pragma unroll
    for (int q = 0; q < 4; ++q) {
        const float bv = bias[colB0 + (wn * 4 + q) * 16 + n];
        #pragma unroll
        for (int r = 0; r < 8; ++r) acc[q][r] = bv;
    }

    // Fixed per-thread staging coordinates
    const int arow = tid >> 3;          // 0..31 : row within block tile
    const int ak0  = (tid & 7) * 4;     // 0..28 : k start (float4 along K)
    const int albuf = arow >> 4;        // which wm sub-buffer
    const int algrp = ak0 >> 4;         // lane group for this k quad
    const int ae0   = ak0 & 15;
    const int al    = (arow & 15) | (algrp << 4);

    auto stage = [&](int buf, int kbase) {
        // A: 32 rows x 32 k, one float4 along K per thread, packed b64 store
        {
            const v4f xv = *reinterpret_cast<const v4f*>(
                &X[(size_t)(row0 + arow) * K + kbase + ak0]);
            unsigned p0 = bf_bits(xv[0]) | (bf_bits(xv[1]) << 16);
            unsigned p1 = bf_bits(xv[2]) | (bf_bits(xv[3]) << 16);
            uint2 pk; pk.x = p0; pk.y = p1;
            *reinterpret_cast<uint2*>(&Afrag[buf][albuf][al][ae0]) = pk;
        }
        // B: 32 k x 256 cols, 8 float4 along cols per thread
        #pragma unroll
        for (int f = 0; f < 8; ++f) {
            const int fid = f * 256 + tid;        // 0..2047 float4 tiles
            const int k   = fid >> 6;             // 0..31
            const int c0  = (fid & 63) * 4;       // 0..252
            const v4f wv = *reinterpret_cast<const v4f*>(
                &W[(size_t)(kbase + k) * N + colB0 + c0]);
            const int e = k & 15;
            #pragma unroll
            for (int q = 0; q < 4; ++q) {
                const int c = c0 + q;
                const int l = (c & 15) | (k & 16);
                Bfrag[buf][c >> 4][l][e] = f2bf(wv[q]);
            }
        }
    };

    const int nIter = K >> 5;           // K / 32
    stage(0, 0);

    for (int it = 0; it < nIter; ++it) {
        const int buf = it & 1;
        __syncthreads();                // publish buf; protect buf^1 rewrite

        if (it + 1 < nIter)             // stage next tile under this WMMA set
            stage(buf ^ 1, (it + 1) << 5);
        if (it + 2 < nIter)             // pull tile after next toward L2
            __builtin_prefetch(
                &X[(size_t)(row0 + arow) * K + ((it + 2) << 5) + ak0], 0, 2);

        // All fragment loads first, then the 4-WMMA clause (no per-WMMA wait;
        // shared A source, disjoint accumulators -> no data hazards).
        const v16bf a  = *reinterpret_cast<const v16bf*>(&Afrag[buf][wm][lane][0]);
        const v16bf b0 = *reinterpret_cast<const v16bf*>(&Bfrag[buf][wn * 4 + 0][lane][0]);
        const v16bf b1 = *reinterpret_cast<const v16bf*>(&Bfrag[buf][wn * 4 + 1][lane][0]);
        const v16bf b2 = *reinterpret_cast<const v16bf*>(&Bfrag[buf][wn * 4 + 2][lane][0]);
        const v16bf b3 = *reinterpret_cast<const v16bf*>(&Bfrag[buf][wn * 4 + 3][lane][0]);
        acc[0] = __builtin_amdgcn_wmma_f32_16x16x32_bf16(false, a, false, b0, (short)0, acc[0], false, false);
        acc[1] = __builtin_amdgcn_wmma_f32_16x16x32_bf16(false, a, false, b1, (short)0, acc[1], false, false);
        acc[2] = __builtin_amdgcn_wmma_f32_16x16x32_bf16(false, a, false, b2, (short)0, acc[2], false, false);
        acc[3] = __builtin_amdgcn_wmma_f32_16x16x32_bf16(false, a, false, b3, (short)0, acc[3], false, false);
    }

    // C/D layout: VGPR r -> M = r (lanes 0-15) / r+8 (lanes 16-31)
    #pragma unroll
    for (int q = 0; q < 4; ++q) {
        const int col = colB0 + (wn * 4 + q) * 16 + n;
        #pragma unroll
        for (int r = 0; r < 8; ++r) {
            const int m = row0 + wm * 16 + r + (hi ? 8 : 0);
            O[(size_t)m * N + col] = acc[q][r];
        }
    }
}

// ---------------------------------------------------------------------------
// pos_feat[32][16] = rel_pos_emb[32][16] @ W_pos[16][16]  (token independent)
// ---------------------------------------------------------------------------
__global__ void posfeat_kernel(const float* __restrict__ rel,
                               const float* __restrict__ Wpos,
                               float* __restrict__ pf)
{
    int i = threadIdx.x;            // 512 threads: j = i>>4, o = i&15
    if (i < DELTA_ * BD_) {
        int j = i >> 4, o = i & 15;
        float f = 0.f;
        #pragma unroll
        for (int k = 0; k < BD_; ++k) f += rel[j * BD_ + k] * Wpos[k * BD_ + o];
        pf[i] = f;
    }
}

// ---------------------------------------------------------------------------
// Windowed bond/damage attention: one wave32 per (b, nh, t); lane = window j.
// Small-K (16) irregular work -> VALU + wave32 shuffles is the right unit.
// ---------------------------------------------------------------------------
__global__ __launch_bounds__(256) void bond_attn_kernel(
        const float* __restrict__ disp,
        const float* __restrict__ val,
        const float* __restrict__ pf,
        const float* __restrict__ W_strain,
        const float* __restrict__ b_strain,
        const float* __restrict__ W_bond,
        const float* __restrict__ b_bond,
        const float* __restrict__ W_dmg,
        const float* __restrict__ b_dmg,
        const float* __restrict__ W_dout,
        const float* __restrict__ b_dout,
        float* __restrict__ out)
{
    __shared__ float sWs[BD_ * BD_], sWd[BD_ * BD_], sPos[DELTA_ * BD_];
    __shared__ float sbs[BD_], sbd[BD_], sWb[BD_], sWo[BD_];

    const int tid = threadIdx.x;
    sWs[tid]        = W_strain[tid];
    sWd[tid]        = W_dmg[tid];
    sPos[tid]       = pf[tid];
    sPos[tid + 256] = pf[tid + 256];
    if (tid < BD_) {
        sbs[tid] = b_strain[tid];
        sbd[tid] = b_dmg[tid];
        sWb[tid] = W_bond[tid];   // (BD,1) flattened
        sWo[tid] = W_dout[tid];
    }
    __syncthreads();

    const float bbond = b_bond[0];
    const float bdout = b_dout[0];

    const int lane = tid & 31;
    const int wid  = blockIdx.x * 8 + (tid >> 5);   // (b*NH + nh)*T + t
    const int t  = wid & (T_ - 1);
    const int nh = (wid >> 11) & (NH_ - 1);
    const int b  = wid >> 15;

    const int  j     = lane;               // window slot
    const int  p     = t - (DELTA_ - 1) + j;
    const bool valid = p >= 0;

    // strain = disp[neighbor] - disp[center]  (neighbor zero-padded)
    float s[BD_];
    {
        const float* dc = disp + ((size_t)(b * T_ + t)) * (NH_ * BD_) + nh * BD_;
        #pragma unroll
        for (int k = 0; k < BD_; ++k) s[k] = -dc[k];
        if (valid) {
            const float* dn = disp + ((size_t)(b * T_ + p)) * (NH_ * BD_) + nh * BD_;
            #pragma unroll
            for (int k = 0; k < BD_; ++k) s[k] += dn[k];
        }
    }

    // bond logit and damage
    float logit;
    {
        float bl = 0.f, dl = 0.f;
        #pragma unroll
        for (int o = 0; o < BD_; ++o) {
            float fb = sbs[o] + sPos[j * BD_ + o];
            float fd = sbd[o];
            #pragma unroll
            for (int k = 0; k < BD_; ++k) {
                fb += s[k] * sWs[k * BD_ + o];
                fd += s[k] * sWd[k * BD_ + o];
            }
            bl += gelu_erf(fb) * sWb[o];
            dl += gelu_erf(fd) * sWo[o];
        }
        float dmg = 1.f / (1.f + __expf(-(dl + bdout)));
        logit = (bl + bbond) - 10.f * dmg;
        if (!valid) logit = -1e30f;        // masked lane
    }

    // softmax across the 32 lanes (wave32: explicit width 32)
    float mx = logit;
    #pragma unroll
    for (int off = 16; off; off >>= 1) mx = fmaxf(mx, __shfl_xor(mx, off, 32));
    float e = valid ? __expf(logit - mx) : 0.f;
    float sum = e;
    #pragma unroll
    for (int off = 16; off; off >>= 1) sum += __shfl_xor(sum, off, 32);
    const float w = e / sum;

    // out[hs] = sum_j w_j * val[neighbor_j][hs]; lane handles hs = lane, lane+32
    float o0 = 0.f, o1 = 0.f;
    for (int jj = 0; jj < DELTA_; ++jj) {
        float wj = __shfl(w, jj, 32);
        int  pp  = t - (DELTA_ - 1) + jj;
        if (pp >= 0) {
            const float* vp = val + ((size_t)(b * T_ + pp)) * (NH_ * HS_) + nh * HS_;
            o0 += wj * vp[lane];
            o1 += wj * vp[lane + 32];
        }
    }
    float* op = out + ((size_t)(b * T_ + t)) * (NH_ * HS_) + nh * HS_;
    op[lane]      = o0;
    op[lane + 32] = o1;
}

// ---------------------------------------------------------------------------
extern "C" void kernel_launch(void* const* d_in, const int* in_sizes, int n_in,
                              void* d_out, int out_size, void* d_ws, size_t ws_size,
                              hipStream_t stream)
{
    (void)in_sizes; (void)n_in; (void)out_size; (void)ws_size;

    const float* x        = (const float*)d_in[0];
    const float* W_disp   = (const float*)d_in[1];
    const float* b_disp   = (const float*)d_in[2];
    const float* W_val    = (const float*)d_in[3];
    const float* b_val    = (const float*)d_in[4];
    const float* rel_pos  = (const float*)d_in[5];
    const float* W_strain = (const float*)d_in[6];
    const float* b_strain = (const float*)d_in[7];
    const float* W_pos    = (const float*)d_in[8];
    const float* W_bond   = (const float*)d_in[9];
    const float* b_bond   = (const float*)d_in[10];
    const float* W_dmg    = (const float*)d_in[11];
    const float* b_dmg    = (const float*)d_in[12];
    const float* W_dout   = (const float*)d_in[13];
    const float* b_dout   = (const float*)d_in[14];
    const float* W_cproj  = (const float*)d_in[15];
    const float* b_cproj  = (const float*)d_in[16];
    float* out = (float*)d_out;

    // Workspace layout (floats): disp (4096x256) | val (4096x1024) |
    //                            attn (4096x1024) | posfeat (32x16)
    float* disp = (float*)d_ws;
    float* val  = disp + (size_t)M_ * (NH_ * BD_);
    float* attn = val  + (size_t)M_ * C_;
    float* pf   = attn + (size_t)M_ * C_;

    // 1) token-independent positional features
    posfeat_kernel<<<1, 512, 0, stream>>>(rel_pos, W_pos, pf);

    // 2) disp = x @ W_disp + b   (4096 x 1024 x 256)
    wmma_gemm_bias<<<dim3(M_ / 32, (NH_ * BD_) / 256), 256, 0, stream>>>(
        x, W_disp, b_disp, disp, M_, NH_ * BD_, C_);

    // 3) val = x @ W_val + b     (4096 x 1024 x 1024)
    wmma_gemm_bias<<<dim3(M_ / 32, C_ / 256), 256, 0, stream>>>(
        x, W_val, b_val, val, M_, C_, C_);

    // 4) windowed bond/damage attention: one wave per (b, nh, t)
    bond_attn_kernel<<<(B_ * NH_ * T_) / 8, 256, 0, stream>>>(
        disp, val, pf, W_strain, b_strain, W_bond, b_bond,
        W_dmg, b_dmg, W_dout, b_dout, attn);

    // 5) out = attn @ W_cproj + b  (4096 x 1024 x 1024)
    wmma_gemm_bias<<<dim3(M_ / 32, C_ / 256), 256, 0, stream>>>(
        attn, W_cproj, b_cproj, out, M_, C_, C_);
}
